// RepresentationEncoderState_6743098655681
// MI455X (gfx1250) — matrix operand order repaired and checked by reading, not compiled, not run on hardware
//
#include <hip/hip_runtime.h>
#include <hip/hip_bf16.h>

// Problem constants (from reference): B=64, T=1024, I=256, H=128, 4H=512
#define B_   64
#define T_   1024
#define I_   256
#define H_   128
#define G4   512
#define ZONEOUT_F 0.15f

typedef __attribute__((ext_vector_type(16))) __bf16 v16bf;
typedef __attribute__((ext_vector_type(8)))  float  v8f;

union Frag16 { uint4 u[2]; v16bf v; };          // 32B = 16 bf16 = one WMMA A/B fragment per lane

__device__ __forceinline__ unsigned short f32_to_bf16(float f) {
    union { float f; unsigned u; } x; x.f = f;
    unsigned r = x.u + 0x7FFFu + ((x.u >> 16) & 1u);   // round-to-nearest-even
    return (unsigned short)(r >> 16);
}

__device__ __forceinline__ float fast_sigmoid(float x) {
    return __builtin_amdgcn_rcpf(1.f + __expf(-x));
}
__device__ __forceinline__ float fast_tanh(float x) {
    // tanh(x) = 1 - 2/(exp(2x)+1); branch-free, correct limits at +/-inf
    return 1.f - 2.f * __builtin_amdgcn_rcpf(__expf(2.f * x) + 1.f);
}

// ---------------------------------------------------------------------------
// Kernel 0a: x [B][T][I] f32  ->  xb [T][B][I] bf16 (rows of the big GEMM)
// ---------------------------------------------------------------------------
__global__ __launch_bounds__(256) void pack_x(const float* __restrict__ x,
                                              unsigned short* __restrict__ xb) {
    size_t tid = (size_t)blockIdx.x * blockDim.x + threadIdx.x;   // B*T*I/4 threads
    size_t lin = tid * 4;                                         // linear in x layout
    int    k   = (int)(lin % I_);
    int    t   = (int)((lin / I_) % T_);
    int    b   = (int)(lin / ((size_t)T_ * I_));
    float4 v   = *(const float4*)(x + lin);
    ushort4 o;
    o.x = f32_to_bf16(v.x); o.y = f32_to_bf16(v.y);
    o.z = f32_to_bf16(v.z); o.w = f32_to_bf16(v.w);
    size_t dst = ((size_t)t * B_ + b) * I_ + k;
    *(ushort4*)(xb + dst) = o;
}

// ---------------------------------------------------------------------------
// Kernel 0b: Wx [256][512] -> WxT [512][256] bf16 ;  Wh [128][512] -> WhT [512][128] bf16
// (transposed so each WMMA B-fragment is a contiguous 32-byte K run)
// ---------------------------------------------------------------------------
__global__ __launch_bounds__(256) void pack_wT(const float* __restrict__ Wx,
                                               const float* __restrict__ Wh,
                                               unsigned short* __restrict__ WxT,
                                               unsigned short* __restrict__ WhT) {
    int tid = blockIdx.x * blockDim.x + threadIdx.x;
    if (tid < G4 * I_) {
        int n = tid / I_, k = tid % I_;
        WxT[tid] = f32_to_bf16(Wx[(size_t)k * G4 + n]);
    }
    if (tid < G4 * H_) {
        int n = tid / H_, k = tid % H_;
        WhT[tid] = f32_to_bf16(Wh[(size_t)k * G4 + n]);
    }
}

// ---------------------------------------------------------------------------
// Kernel 1: Gx[t*B+b][n] = sum_k xb[t*B+b][k] * Wx[k][n]
// M=65536, N=512, K=256. 8 waves/block; wave = 32(M) x 64(N) -> each B-fragment
// feeds two WMMAs (register blocking over M). Block = 256(M) x 64(N).
// ---------------------------------------------------------------------------
__global__ __launch_bounds__(256) void gemm_xwx(const unsigned short* __restrict__ xb,
                                                const unsigned short* __restrict__ WxT,
                                                float* __restrict__ Gx) {
    const int lane = threadIdx.x & 31;
    const int w    = threadIdx.x >> 5;          // 0..7
    const int row  = lane & 15;
    const int half = lane >> 4;
    const int bn   = blockIdx.x & 7;            // 8 N-blocks of 64
    const int bm   = blockIdx.x >> 3;           // 256 M-blocks of 256
    const int m0   = (bm * 8 + w) * 32;         // wave covers rows [m0, m0+32)
    const int n0b  = bn * 64;

    v8f c[2][4] = {};
    const unsigned short* arow0 = xb + (size_t)(m0 + row) * I_;
    const unsigned short* arow1 = xb + (size_t)(m0 + 16 + row) * I_;

    for (int kt = 0; kt < 8; ++kt) {            // K = 8 x 32
        const int kk = kt * 32 + half * 16;
        Frag16 a0, a1;
        a0.u[0] = *(const uint4*)(arow0 + kk);
        a0.u[1] = *(const uint4*)(arow0 + kk + 8);
        a1.u[0] = *(const uint4*)(arow1 + kk);
        a1.u[1] = *(const uint4*)(arow1 + kk + 8);
#pragma unroll
        for (int nt = 0; nt < 4; ++nt) {
            const unsigned short* bp =
                WxT + (size_t)(n0b + nt * 16 + row) * I_ + kt * 32 + half * 16;
            Frag16 bf;
            bf.u[0] = *(const uint4*)(bp);
            bf.u[1] = *(const uint4*)(bp + 8);
            c[0][nt] = __builtin_amdgcn_wmma_f32_16x16x32_bf16(
                           false, a0.v, false, bf.v, (short)0, c[0][nt], false, false);
            c[1][nt] = __builtin_amdgcn_wmma_f32_16x16x32_bf16(
                           false, a1.v, false, bf.v, (short)0, c[1][nt], false, false);
        }
    }
#pragma unroll
    for (int mt = 0; mt < 2; ++mt)
#pragma unroll
        for (int nt = 0; nt < 4; ++nt) {
            float* dst = Gx + (size_t)(m0 + mt * 16 + 8 * half) * G4 + n0b + nt * 16 + row;
#pragma unroll
            for (int v = 0; v < 8; ++v) dst[(size_t)v * G4] = c[mt][nt][v];  // D: M = v + 8*half
        }
}

// ---------------------------------------------------------------------------
// Kernel 2: persistent single-WGP recurrence.
// 512 threads = 16 waves. Wave w owns gate columns [32w, 32w+32): Wh fragments
// stay resident in VGPRs; h (bf16) lives in LDS; gates exchanged via LDS;
// cell / hidden-prev state lives in per-thread registers. Next timestep's Gx
// block is prefetched (global_prefetch_b8) to pull HBM latency off the
// serial dependence chain.
// ---------------------------------------------------------------------------
__global__ __launch_bounds__(512) void lstm_recurrent(const float* __restrict__ Gx,
                                                      const unsigned short* __restrict__ WhT,
                                                      const float* __restrict__ bias,
                                                      float* __restrict__ hids,
                                                      float* __restrict__ cels,
                                                      float* __restrict__ pogs) {
    extern __shared__ unsigned char smem[];
    float*          gates = (float*)smem;                                   // [64][512] f32 (128 KB)
    unsigned short* hid   = (unsigned short*)(smem + (size_t)B_ * G4 * 4);  // [64][128] bf16 (16 KB)

    const int tid  = threadIdx.x;
    const int lane = tid & 31;
    const int w    = tid >> 5;              // 0..15
    const int row  = lane & 15;
    const int half = lane >> 4;
    const int n0   = w * 32;                // this wave's N strip

    // Resident Wh B-fragments: [nt][kt], nt in 0..1 (N), kt in 0..3 (K=128)
    Frag16 bfrag[2][4];
#pragma unroll
    for (int nt = 0; nt < 2; ++nt)
#pragma unroll
        for (int kt = 0; kt < 4; ++kt) {
            const unsigned short* p =
                WhT + (size_t)(n0 + nt * 16 + row) * H_ + kt * 32 + half * 16;
            bfrag[nt][kt].u[0] = *(const uint4*)p;
            bfrag[nt][kt].u[1] = *(const uint4*)(p + 8);
        }
    float bv[2];
    bv[0] = bias[n0 + row];
    bv[1] = bias[n0 + 16 + row];

    // Elementwise mapping: thread -> (batch eb, 16 h values starting at eh)
    const int eb = tid >> 3;
    const int eh = (tid & 7) * 16;
    float cel[16], hprev[16];
#pragma unroll
    for (int j = 0; j < 16; ++j) { cel[j] = 0.f; hprev[j] = 0.f; }

    for (int i = tid; i < B_ * H_; i += 512) hid[i] = 0;   // h0 = 0
    __syncthreads();

    const float z = ZONEOUT_F, omz = 1.f - ZONEOUT_F;

    for (int t = 0; t < T_; ++t) {
        const float* gxt = Gx + (size_t)t * (B_ * G4);

        // Prefetch next step's 128 KB Gx block: 512 thr x 2 lines x 128 B.
        if (t + 1 < T_) {
            const float* nxt = gxt + (B_ * G4);
            __builtin_prefetch(nxt + (size_t)tid * 32, 0, 0);
            __builtin_prefetch(nxt + (size_t)(512 + tid) * 32, 0, 0);
        }

        // ---- GEMM phase: gates = Gx + b + h @ Wh (bf16 WMMA, f32 accum) ----
        for (int mt = 0; mt < 4; ++mt) {
            Frag16 a[4];
            const unsigned short* ar = hid + (mt * 16 + row) * H_;
#pragma unroll
            for (int kt = 0; kt < 4; ++kt) {
                const int kk = kt * 32 + half * 16;
                a[kt].u[0] = *(const uint4*)(ar + kk);
                a[kt].u[1] = *(const uint4*)(ar + kk + 8);
            }
#pragma unroll
            for (int nt = 0; nt < 2; ++nt) {
                const int nc = n0 + nt * 16 + row;
                const float* gsrc = gxt + (size_t)(mt * 16 + 8 * half) * G4 + nc;
                v8f c;
#pragma unroll
                for (int v = 0; v < 8; ++v) c[v] = gsrc[v * G4] + bv[nt];
#pragma unroll
                for (int kt = 0; kt < 4; ++kt)
                    c = __builtin_amdgcn_wmma_f32_16x16x32_bf16(
                            false, a[kt].v, false, bfrag[nt][kt].v, (short)0, c, false, false);
                float* gdst = gates + (mt * 16 + 8 * half) * G4 + nc;
#pragma unroll
                for (int v = 0; v < 8; ++v) gdst[v * G4] = c[v];
            }
        }
        __syncthreads();

        // ---- Elementwise phase: sigmoid/tanh, cell update, zoneout ----
        const float* gr = gates + eb * G4;        // [i | f | g | o], 128 each
        float ga[4][16];
#pragma unroll
        for (int g = 0; g < 4; ++g)
#pragma unroll
            for (int q = 0; q < 4; ++q)
                *(float4*)&ga[g][4 * q] = *(const float4*)(gr + g * H_ + eh + 4 * q);

        float hout[16], cout_[16], pout[16];
        unsigned short hb[16];
#pragma unroll
        for (int j = 0; j < 16; ++j) {
            float ig = fast_sigmoid(ga[0][j]);
            float fg = fast_sigmoid(ga[1][j]);
            float gg = fast_tanh(ga[2][j]);
            float og = fast_sigmoid(ga[3][j]);
            float cn = fg * cel[j] + ig * gg;
            float hn = og * fast_tanh(cn);
            float cz = z * cel[j]   + omz * cn;   // zoneout-interpolated cell (carried)
            float hz = z * hprev[j] + omz * hn;   // zoneout-interpolated hidden (carried)
            cel[j] = cz; hprev[j] = hz;
            hout[j] = hz; cout_[j] = cz; pout[j] = og;
            hb[j] = f32_to_bf16(hz);
        }

        const size_t obase = ((size_t)eb * T_ + t) * H_ + eh;
#pragma unroll
        for (int q = 0; q < 4; ++q) {
            *(float4*)(hids + obase + 4 * q) = *(float4*)&hout[4 * q];
            *(float4*)(cels + obase + 4 * q) = *(float4*)&cout_[4 * q];
            *(float4*)(pogs + obase + 4 * q) = *(float4*)&pout[4 * q];
        }

        union { unsigned short s[16]; uint4 u[2]; } hu;
#pragma unroll
        for (int j = 0; j < 16; ++j) hu.s[j] = hb[j];
        *(uint4*)(hid + eb * H_ + eh)     = hu.u[0];
        *(uint4*)(hid + eb * H_ + eh + 8) = hu.u[1];
        __syncthreads();
    }
}

// ---------------------------------------------------------------------------
extern "C" void kernel_launch(void* const* d_in, const int* in_sizes, int n_in,
                              void* d_out, int out_size, void* d_ws, size_t ws_size,
                              hipStream_t stream) {
    const float* x  = (const float*)d_in[0];   // [B,T,I]
    const float* Wx = (const float*)d_in[1];   // [I,4H]
    const float* Wh = (const float*)d_in[2];   // [H,4H]
    const float* b  = (const float*)d_in[3];   // [4H]

    unsigned char*  ws  = (unsigned char*)d_ws;
    unsigned short* xb  = (unsigned short*)(ws);                // 32 MiB : [T*B][I] bf16
    unsigned short* WxT = (unsigned short*)(ws + 33554432u);    // 256 KiB: [512][256] bf16
    unsigned short* WhT = (unsigned short*)(ws + 33816576u);    // 128 KiB: [512][128] bf16
    float*          Gx  = (float*)(ws + 33947648u);             // 128 MiB: [T*B][512] f32

    float* hids = (float*)d_out;
    float* cels = hids + (size_t)B_ * T_ * H_;
    float* pogs = cels + (size_t)B_ * T_ * H_;

    pack_x  <<<(B_ * T_ * I_ / 4 + 255) / 256, 256, 0, stream>>>(x, xb);
    pack_wT <<<(G4 * I_ + 255) / 256,          256, 0, stream>>>(Wx, Wh, WxT, WhT);
    gemm_xwx<<<2048,                           256, 0, stream>>>(xb, WxT, Gx);

    const size_t lds_bytes = (size_t)B_ * G4 * 4 + (size_t)B_ * H_ * 2;  // 147456 B
    lstm_recurrent<<<1, 512, lds_bytes, stream>>>(Gx, WhT, b, hids, cels, pogs);
}
